// DKVMN_47785806135738
// MI455X (gfx1250) — matrix operand order repaired and checked by reading, not compiled, not run on hardware
//
#include <hip/hip_runtime.h>

// ---------------- problem constants ----------------
#define B_   64
#define T_   200
#define D_   256
#define M_   64
#define NS_  1000
#define ROWS (B_ * T_)            // 12800
#define NTILES (ROWS / 16)        // 800 row tiles

typedef __attribute__((ext_vector_type(16))) __bf16 v16bf;
typedef __attribute__((ext_vector_type(8)))  float  v8f;

// fp32 -> bf16 (round to nearest even)
__device__ __forceinline__ unsigned short f2bf(float f) {
  union { float f; unsigned int u; } c; c.f = f;
  unsigned int u = c.u;
  u += 0x7fffu + ((u >> 16) & 1u);
  return (unsigned short)(u >> 16);
}

// A fragment: 16x32 bf16 tile (rows = M, cols = K), tile row-major (ld = D_)
// ISA layout: lanes 0-15 -> M=lane, VGPR j: K = {2j,2j+1} (j<4) / {16+2(j-4),...} (j>=4)
//             lanes 16-31 -> same M, K offset by +8
__device__ __forceinline__ v16bf load_a_frag(const unsigned short* tile, int lane, int k0) {
  const int mrow = lane & 15, half = lane >> 4;
  union { v16bf v; unsigned int u[8]; } f;
  const unsigned short* p = tile + mrow * D_ + k0 + half * 8;
#pragma unroll
  for (int j = 0; j < 8; ++j) {
    const int kl = (j < 4) ? (2 * j) : (16 + 2 * (j - 4));
    f.u[j] = *(const unsigned int*)(p + kl);
  }
  return f.v;
}

// B fragment: 32x16 bf16 (K x N).  B[k][n] = W[n][k], W row-major [N][ldw].
// Inferred layout: lane = N (0-15) / N (16-31 => K+8), VGPR j: K = khalf*8 + {2j,2j+1} (j<4) / khalf*8 + {16+2(j-4),...} (j>=4)
__device__ __forceinline__ v16bf load_b_frag(const unsigned short* __restrict__ W, int ldw,
                                             int n0, int k0, int lane) {
  const int col = lane & 15, khalf = lane >> 4;
  const unsigned short* p = W + (size_t)(n0 + col) * ldw + k0 + khalf * 8;
  union { v16bf v; unsigned int u[8]; } f;
#pragma unroll
  for (int j = 0; j < 8; ++j) f.u[j] = *(const unsigned int*)(p + ((j < 4) ? (2 * j) : (16 + 2 * (j - 4))));
  return f.v;
}

// ---------------- kernel 0: fp32 -> bf16 table conversion ----------------
__global__ __launch_bounds__(256) void dkvmn_cvt_bf16(const float* __restrict__ src,
                                                      unsigned short* __restrict__ dst, int n) {
  int i = blockIdx.x * 256 + threadIdx.x;
  if (i < n) dst[i] = f2bf(src[i]);
}

// ---------------- kernel 1: gather + all t-parallel GEMMs + softmax ----------------
// Per block: one 16-row tile of (b,t).  8 waves share 52 WMMA jobs:
//  jobs 0-3   : logits = k . Mk^T       (64 cols)  -> LDS, softmax -> w
//  jobs 4-19  : e = sigmoid(v . eW^T+eb)           -> global
//  jobs 20-35 : a = tanh(v . aW^T+ab)              -> global
//  jobs 36-51 : kf = k . fW[:,256:]^T + fb         -> global
__global__ __launch_bounds__(256) void dkvmn_gates(
    const int* __restrict__ skills, const int* __restrict__ responses,
    const unsigned short* __restrict__ kemb_bf, const unsigned short* __restrict__ vemb_bf,
    const unsigned short* __restrict__ Mk_bf, const unsigned short* __restrict__ eW_bf,
    const unsigned short* __restrict__ aW_bf, const unsigned short* __restrict__ fW_bf,
    const float* __restrict__ eb, const float* __restrict__ ab, const float* __restrict__ fb,
    float* __restrict__ w_ws, float* __restrict__ e_ws,
    float* __restrict__ a_ws, float* __restrict__ kf_ws) {
  __shared__ unsigned short k_tile[16 * D_];
  __shared__ unsigned short v_tile[16 * D_];
  __shared__ float logit[16 * M_];

  const int tid = threadIdx.x;
  const int row0 = blockIdx.x * 16;

  // gather 16 k-rows and v-rows (bf16) into LDS, dword-wide
  for (int i = tid; i < 16 * (D_ / 2); i += 256) {
    const int r = i >> 7;          // local row
    const int c = i & 127;         // dword column
    const int g = row0 + r;        // flat (b*T + t)
    const int sk = skills[g];
    const int xr = sk + NS_ * responses[g];
    ((unsigned int*)k_tile)[r * 128 + c] = ((const unsigned int*)(kemb_bf + (size_t)sk * D_))[c];
    ((unsigned int*)v_tile)[r * 128 + c] = ((const unsigned int*)(vemb_bf + (size_t)xr * D_))[c];
  }
  __syncthreads();

  const int wave = tid >> 5, lane = tid & 31;
  for (int job = wave; job < 52; job += 8) {
    const unsigned short* src; const unsigned short* W;
    int ldw, n0, koff = 0, type;
    if (job < 4)       { type = 0; src = k_tile; W = Mk_bf; ldw = D_;     n0 = job * 16; }
    else if (job < 20) { type = 1; src = v_tile; W = eW_bf; ldw = D_;     n0 = (job - 4) * 16; }
    else if (job < 36) { type = 2; src = v_tile; W = aW_bf; ldw = D_;     n0 = (job - 20) * 16; }
    else               { type = 3; src = k_tile; W = fW_bf; ldw = 2 * D_; n0 = (job - 36) * 16; koff = D_; }

    v8f acc = {0.f, 0.f, 0.f, 0.f, 0.f, 0.f, 0.f, 0.f};
#pragma unroll
    for (int kc = 0; kc < 8; ++kc) {
      v16bf af = load_a_frag(src, lane, kc * 32);
      v16bf bf = load_b_frag(W, ldw, n0, koff + kc * 32, lane);
      acc = __builtin_amdgcn_wmma_f32_16x16x32_bf16(false, af, false, bf, (short)0, acc,
                                                    false, false);
    }
    const int colN = lane & 15, half = lane >> 4;
#pragma unroll
    for (int r = 0; r < 8; ++r) {
      const int Mrow = r + 8 * half;
      const int g = row0 + Mrow;
      const int n = n0 + colN;
      float val = acc[r];
      if (type == 0) {
        logit[Mrow * M_ + n] = val;
      } else if (type == 1) {
        e_ws[(size_t)g * D_ + n] = 1.0f / (1.0f + __expf(-(val + eb[n])));
      } else if (type == 2) {
        a_ws[(size_t)g * D_ + n] = tanhf(val + ab[n]);
      } else {
        kf_ws[(size_t)g * D_ + n] = val + fb[n];
      }
    }
  }
  __syncthreads();

  // softmax over the 64 memory slots, one thread per row
  if (tid < 16) {
    const int g = row0 + tid;
    float mx = -1e30f;
    for (int m = 0; m < M_; ++m) mx = fmaxf(mx, logit[tid * M_ + m]);
    float s = 0.f;
    for (int m = 0; m < M_; ++m) s += __expf(logit[tid * M_ + m] - mx);
    const float inv = 1.0f / s;
    for (int m = 0; m < M_; ++m)
      w_ws[(size_t)g * M_ + m] = __expf(logit[tid * M_ + m] - mx) * inv;
  }
}

// ---------------- kernel 2: serial scan, one block per batch ----------------
// Thread d owns column d of Mv: the whole 64-float state column lives in VGPRs.
// w broadcast via wave32 shuffles -> no LDS, no barriers, pure VALU recurrence.
__global__ __launch_bounds__(256) void dkvmn_scan(
    const float* __restrict__ w_ws, const float* __restrict__ e_ws,
    const float* __restrict__ a_ws, const float* __restrict__ Mv0,
    float* __restrict__ reads_ws) {
  const int b = blockIdx.x, tid = threadIdx.x, lane = tid & 31;
  float mv[M_];
#pragma unroll
  for (int m = 0; m < M_; ++m) mv[m] = Mv0[m * D_ + tid];

  for (int t = 0; t < T_; ++t) {
    const size_t base = (size_t)(b * T_ + t);
    const float w0 = w_ws[base * M_ + lane];
    const float w1 = w_ws[base * M_ + 32 + lane];
    const float ed = e_ws[base * D_ + tid];
    const float ad = a_ws[base * D_ + tid];
    float rd = 0.f;
#pragma unroll
    for (int m = 0; m < M_; ++m) {
      const float wm = (m < 32) ? __shfl(w0, m, 32) : __shfl(w1, m - 32, 32);
      const float old = mv[m];
      rd += wm * old;                             // read BEFORE write
      mv[m] = old * (1.0f - wm * ed) + wm * ad;   // erase + add
    }
    reads_ws[base * D_ + tid] = rd;
  }
}

// ---------------- kernel 3: f = tanh(kf + reads.fW_r^T), p = sigmoid(f.pW+pb) ----------------
__global__ __launch_bounds__(256) void dkvmn_out(
    const float* __restrict__ reads_ws, const unsigned short* __restrict__ fW_bf,
    const float* __restrict__ kf_ws, const float* __restrict__ pW,
    const float* __restrict__ pb, float* __restrict__ out) {
  __shared__ unsigned short r_tile[16 * D_];
  __shared__ float f_tile[16 * D_];
  __shared__ float partial[16 * 16];

  const int tid = threadIdx.x;
  const int row0 = blockIdx.x * 16;

  for (int i = tid; i < 16 * D_; i += 256) {
    const int r = i >> 8, c = i & 255;
    r_tile[i] = f2bf(reads_ws[(size_t)(row0 + r) * D_ + c]);
  }
  __syncthreads();

  const int wave = tid >> 5, lane = tid & 31;
  for (int job = wave; job < 16; job += 8) {
    const int n0 = job * 16;
    v8f acc = {0.f, 0.f, 0.f, 0.f, 0.f, 0.f, 0.f, 0.f};
#pragma unroll
    for (int kc = 0; kc < 8; ++kc) {
      v16bf af = load_a_frag(r_tile, lane, kc * 32);
      v16bf bf = load_b_frag(fW_bf, 2 * D_, n0, kc * 32, lane);  // fW[:, 0:256]
      acc = __builtin_amdgcn_wmma_f32_16x16x32_bf16(false, af, false, bf, (short)0, acc,
                                                    false, false);
    }
    const int colN = lane & 15, half = lane >> 4;
#pragma unroll
    for (int r = 0; r < 8; ++r) {
      const int Mrow = r + 8 * half;
      const int g = row0 + Mrow;
      const int n = n0 + colN;
      f_tile[Mrow * D_ + n] = tanhf(acc[r] + kf_ws[(size_t)g * D_ + n]);
    }
  }
  __syncthreads();

  {
    const int row = tid >> 4, seg = tid & 15;
    float s = 0.f;
#pragma unroll
    for (int i = 0; i < 16; ++i) s += f_tile[row * D_ + seg * 16 + i] * pW[seg * 16 + i];
    partial[row * 16 + seg] = s;
  }
  __syncthreads();

  if (tid < 16) {
    float s = pb[0];
#pragma unroll
    for (int i = 0; i < 16; ++i) s += partial[tid * 16 + i];
    const float p = 1.0f / (1.0f + __expf(-s));
    const int g = row0 + tid;
    const int b = g / T_, t = g % T_;
    if (t >= 1) out[(size_t)b * (T_ - 1) + (t - 1)] = p;   // p[:, LENGTH:]
  }
}

// ---------------- host side ----------------
extern "C" void kernel_launch(void* const* d_in, const int* in_sizes, int n_in,
                              void* d_out, int out_size, void* d_ws, size_t ws_size,
                              hipStream_t stream) {
  const int*   skills    = (const int*)d_in[0];
  const int*   responses = (const int*)d_in[1];
  const float* k_emb     = (const float*)d_in[2];
  const float* v_emb     = (const float*)d_in[3];
  const float* Mk        = (const float*)d_in[4];
  const float* Mv0       = (const float*)d_in[5];
  const float* fW        = (const float*)d_in[6];
  const float* fb        = (const float*)d_in[7];
  const float* eW        = (const float*)d_in[8];
  const float* eb        = (const float*)d_in[9];
  const float* aW        = (const float*)d_in[10];
  const float* ab        = (const float*)d_in[11];
  const float* pW        = (const float*)d_in[12];
  const float* pb        = (const float*)d_in[13];
  float* out = (float*)d_out;

  char* wsp = (char*)d_ws;
  auto alloc = [&](size_t bytes) -> char* {
    char* p = wsp;
    wsp += (bytes + 255) & ~(size_t)255;
    return p;
  };
  float* w_ws     = (float*)alloc((size_t)ROWS * M_ * 4);
  float* e_ws     = (float*)alloc((size_t)ROWS * D_ * 4);
  float* a_ws     = (float*)alloc((size_t)ROWS * D_ * 4);
  float* kf_ws    = (float*)alloc((size_t)ROWS * D_ * 4);
  float* reads_ws = (float*)alloc((size_t)ROWS * D_ * 4);
  unsigned short* kemb_bf = (unsigned short*)alloc((size_t)NS_ * D_ * 2);
  unsigned short* vemb_bf = (unsigned short*)alloc((size_t)2 * NS_ * D_ * 2);
  unsigned short* Mk_bf   = (unsigned short*)alloc((size_t)M_ * D_ * 2);
  unsigned short* eW_bf   = (unsigned short*)alloc((size_t)D_ * D_ * 2);
  unsigned short* aW_bf   = (unsigned short*)alloc((size_t)D_ * D_ * 2);
  unsigned short* fW_bf   = (unsigned short*)alloc((size_t)D_ * 2 * D_ * 2);

  auto cvt = [&](const float* s, unsigned short* d, int n) {
    dkvmn_cvt_bf16<<<(n + 255) / 256, 256, 0, stream>>>(s, d, n);
  };
  cvt(k_emb, kemb_bf, NS_ * D_);
  cvt(v_emb, vemb_bf, 2 * NS_ * D_);
  cvt(Mk,    Mk_bf,   M_ * D_);
  cvt(eW,    eW_bf,   D_ * D_);
  cvt(aW,    aW_bf,   D_ * D_);
  cvt(fW,    fW_bf,   D_ * 2 * D_);

  dkvmn_gates<<<NTILES, 256, 0, stream>>>(skills, responses, kemb_bf, vemb_bf,
                                          Mk_bf, eW_bf, aW_bf, fW_bf,
                                          eb, ab, fb, w_ws, e_ws, a_ws, kf_ws);
  dkvmn_scan<<<B_, 256, 0, stream>>>(w_ws, e_ws, a_ws, Mv0, reads_ws);
  dkvmn_out<<<NTILES, 256, 0, stream>>>(reads_ws, fW_bf, kf_ws, pW, pb, out);
}
